// MoDE_66434554135195
// MI455X (gfx1250) — compile-verified, matrix operand order checked by reading, not run on hardware
//
#include <hip/hip_runtime.h>
#include <hip/hip_bf16.h>

typedef __attribute__((ext_vector_type(16))) __bf16        v16bf;
typedef __attribute__((ext_vector_type(8)))  float         v8f;
typedef __attribute__((ext_vector_type(4)))  unsigned int  u32x4;
typedef __attribute__((ext_vector_type(4)))  int           v4i;
typedef __attribute__((ext_vector_type(8)))  int           v8i;

union Frag16 { u32x4 u[2]; v16bf v; };

__device__ __forceinline__ unsigned short f2bf(float f) {
    unsigned int u = __float_as_uint(f);
    unsigned int r = u + 0x7FFFu + ((u >> 16) & 1u);   // round-to-nearest-even
    return (unsigned short)(r >> 16);
}

__device__ __forceinline__ unsigned int lds_offset(const void* p) {
    // LDS aperture: low 32 bits of the generic address are the LDS byte offset.
    return (unsigned int)(unsigned long long)(uintptr_t)p;
}

// ---------------------------------------------------------------------------
// Tensor Data Mover: 2D tile load global->LDS (CDNA5 ISA).
// D# group0: [1:0]=count=1, [63:32]=lds_addr, [120:64]=global_addr, [127:126]=2.
// D# group1: [17:16]=data_size(1=2B), [79:48]=tensor_dim0, [111:80]=tensor_dim1,
//            [127:112]=tile_dim0, [143:128]=tile_dim1, [207:160]=dim0_stride.
// Issued by one wave per block; completion via s_wait_tensorcnt (in-order).
// ---------------------------------------------------------------------------
__device__ __forceinline__ void tdm_load_2d_bf16(
    unsigned int lds_off, unsigned long long ga,
    int tile0, int tile1, long long stride0)
{
    v4i g0;
    g0[0] = 1;                                   // count=1, user mode
    g0[1] = (int)lds_off;                        // lds_addr
    g0[2] = (int)(unsigned int)ga;               // global_addr[31:0]
    g0[3] = (int)(((unsigned int)(ga >> 32)) & 0x01FFFFFFu) |
            (int)0x80000000u;                    // global_addr[56:32] | type=2
    v8i g1;
    g1[0] = 0x00010000;                          // data_size=1 (2 bytes)
    g1[1] = (tile0 & 0xFFFF) << 16;              // tensor_dim0[15:0]
    g1[2] = ((tile0 >> 16) & 0xFFFF) | ((tile1 & 0xFFFF) << 16); // td0 hi | td1 lo
    g1[3] = ((tile1 >> 16) & 0xFFFF) | ((tile0 & 0xFFFF) << 16); // td1 hi | tile_dim0
    g1[4] = (tile1 & 0xFFFF);                    // tile_dim1 (tile_dim2=0)
    g1[5] = (int)(stride0 & 0xFFFFFFFFll);       // tensor_dim0_stride[31:0]
    g1[6] = (int)((stride0 >> 32) & 0xFFFFll);   // tensor_dim0_stride[47:32]
    g1[7] = 0;
    asm volatile("tensor_load_to_lds %0, %1" :: "s"(g0), "s"(g1) : "memory");
}

// ---------------------------------------------------------------------------
// Phase 0: one-time f32 -> bf16 convert + transpose of expert weights.
// src: [e][K][Nn] f32 -> dst: [e][Nn][K] bf16 so GEMM B tiles become
// contiguous [n][k] 2D tiles servable by the TDM with zero VALU work.
// ---------------------------------------------------------------------------
__global__ __launch_bounds__(256) void moe_cvt_transpose(
    const float* __restrict__ src, unsigned short* __restrict__ dst,
    int K, int Nn)
{
    __shared__ unsigned short tile[32][33];
    const int e  = blockIdx.z;
    const int k0 = blockIdx.y * 32, n0 = blockIdx.x * 32;
    const int tid = threadIdx.x;
    const int c = tid & 31, r8 = tid >> 5;
    const float* s = src + ((size_t)e * K + k0) * Nn + n0;
#pragma unroll
    for (int i = 0; i < 4; ++i) {
        const int k = r8 + i * 8;
        tile[k][c] = f2bf(s[(size_t)k * Nn + c]);
    }
    __syncthreads();
    const int n = tid >> 3, ks = (tid & 7) * 4;
    unsigned short* d = dst + ((size_t)e * Nn + n0 + n) * K + k0 + ks;
    const unsigned int lo = (unsigned int)tile[ks][n] | ((unsigned int)tile[ks + 1][n] << 16);
    const unsigned int hi = (unsigned int)tile[ks + 2][n] | ((unsigned int)tile[ks + 3][n] << 16);
    *(uint2*)d = make_uint2(lo, hi);
}

// ---------------------------------------------------------------------------
// Phase 1: router.  One wave per token: 8 dot products, softmax, top-2.
// ---------------------------------------------------------------------------
__global__ __launch_bounds__(256) void moe_router(
    const float* __restrict__ x, const float* __restrict__ rw,
    const float* __restrict__ rb, float* __restrict__ noop_w,
    int* __restrict__ a_e, float* __restrict__ a_w,
    int* __restrict__ blockCounts, int H)
{
    __shared__ int cnt[8];
    const int tid = threadIdx.x;
    if (tid < 8) cnt[tid] = 0;
    __syncthreads();

    const int wave = tid >> 5, lane = tid & 31;
    const int t = blockIdx.x * 8 + wave;
    const float* xr = x + (size_t)t * H;

    float acc[8];
#pragma unroll
    for (int e = 0; e < 8; ++e) acc[e] = 0.f;

    for (int j = lane; j < H; j += 32) {
        const float xv = xr[j];
#pragma unroll
        for (int e = 0; e < 8; ++e) acc[e] += xv * rw[e * H + j];
    }
#pragma unroll
    for (int off = 16; off > 0; off >>= 1) {
#pragma unroll
        for (int e = 0; e < 8; ++e) acc[e] += __shfl_xor(acc[e], off, 32);
    }

    if (lane == 0) {
        float lg[8], w[8];
#pragma unroll
        for (int e = 0; e < 8; ++e) lg[e] = acc[e] + rb[e];
        float m = lg[0];
#pragma unroll
        for (int e = 1; e < 8; ++e) m = fmaxf(m, lg[e]);
        float s = 0.f;
#pragma unroll
        for (int e = 0; e < 8; ++e) { w[e] = __expf(lg[e] - m); s += w[e]; }
        const float inv = 1.f / s;
#pragma unroll
        for (int e = 0; e < 8; ++e) w[e] *= inv;

        int b1 = 0;
#pragma unroll
        for (int e = 1; e < 8; ++e) if (w[e] > w[b1]) b1 = e;
        int b2 = (b1 == 0) ? 1 : 0;
#pragma unroll
        for (int e = 0; e < 8; ++e) if (e != b1 && w[e] > w[b2]) b2 = e;

        noop_w[t] = (b1 == 7 || b2 == 7) ? w[7] : 0.f;
        a_e[2 * t]     = b1;  a_w[2 * t]     = w[b1];
        a_e[2 * t + 1] = b2;  a_w[2 * t + 1] = w[b2];
        if (b1 < 7) atomicAdd(&cnt[b1], 1);
        if (b2 < 7) atomicAdd(&cnt[b2], 1);
    }
    __syncthreads();
    if (tid < 7) blockCounts[blockIdx.x * 7 + tid] = cnt[tid];
}

// ---------------------------------------------------------------------------
// Phase 2: exclusive scan of block histograms + slot-table init.
// ---------------------------------------------------------------------------
__global__ void moe_scan_init(const int* __restrict__ blockCounts,
                              int* __restrict__ blockOffsets,
                              int* __restrict__ slot_token,
                              int nBlocks, int CAP)
{
    const int tid = threadIdx.x;
    for (int i = tid; i < 7 * CAP; i += blockDim.x) slot_token[i] = -1;
    if (tid < 7) {
        int run = 0;
        for (int b = 0; b < nBlocks; ++b) {
            blockOffsets[b * 7 + tid] = run;
            run += blockCounts[b * 7 + tid];
        }
    }
}

// ---------------------------------------------------------------------------
// Phase 3: deterministic slot assignment (matches stable-argsort drop order).
// ---------------------------------------------------------------------------
__global__ void moe_assign(const int* __restrict__ a_e, const float* __restrict__ a_w,
                           const int* __restrict__ blockOffsets,
                           int* __restrict__ slot_token, int* __restrict__ tok_slot,
                           int CAP)
{
    const int w = threadIdx.x;
    if (w >= 8) return;
    const int blk = blockIdx.x;
    const int t = blk * 8 + w;
#pragma unroll
    for (int k = 0; k < 2; ++k) {
        const int e = a_e[2 * t + k];
        int slot = -1;
        if (e < 7) {
            int local = 0;
            for (int t2 = blk * 8; t2 < t; ++t2) {
                if (a_e[2 * t2] == e)     ++local;
                if (a_e[2 * t2 + 1] == e) ++local;
            }
            const int s = blockOffsets[blk * 7 + e] + local;
            if (s < CAP) { slot_token[e * CAP + s] = t; slot = s; }
        }
        tok_slot[2 * t + k] = slot;
    }
}

#define BM 128
#define BN 128
#define BK 32

// ---------------------------------------------------------------------------
// GEMM1: h[e] = relu( gather(x, slot_token[e]) @ W1[e] ), h stored bf16.
// A: VALU gather + f32->bf16.  B: double-buffered TDM tiles — DMA for k+1
// overlaps both A-staging and WMMA of tile k (wait tensorcnt<=1 keeps one
// tile in flight; end-of-iter barrier guarantees the target buffer is free).
// ---------------------------------------------------------------------------
__global__ __launch_bounds__(256) void moe_gemm1(
    const float* __restrict__ x, const unsigned short* __restrict__ w1t,
    const int* __restrict__ slot_token, unsigned short* __restrict__ hbuf,
    int H, int I, int CAP)
{
    __shared__ __align__(16) unsigned short As[BM * BK];       // [m][k]
    __shared__ __align__(16) unsigned short Bs[2][BN * BK];    // [n][k] x2
    __shared__ int stok[BM];

    const int e  = blockIdx.z;
    const int n0 = blockIdx.x * BN;
    const int m0 = blockIdx.y * BM;
    const int tid = threadIdx.x;
    const int wave = tid >> 5, lane = tid & 31;
    const int waveM = wave & 3, waveN = wave >> 2;
    const int lm = lane & 15;
    const int hf = (lane >> 4) * 8;

    if (tid < BM) stok[tid] = slot_token[(size_t)e * CAP + m0 + tid];
    __syncthreads();

    v8f acc[2][4];
#pragma unroll
    for (int i = 0; i < 2; ++i)
#pragma unroll
        for (int j = 0; j < 4; ++j) acc[i][j] = v8f{};

    const int rA = tid >> 1, halfA = tid & 1;
    const unsigned int bOff0 = lds_offset(Bs[0]);
    const unsigned int bOff1 = lds_offset(Bs[1]);
    const unsigned long long bBase =
        (unsigned long long)(uintptr_t)(w1t + ((size_t)e * I + n0) * H);

    const int nIter = H / BK;
    if (wave == 0)                                     // prologue: tile 0
        tdm_load_2d_bf16(bOff0, bBase, BK, BN, H);

    for (int it = 0; it < nIter; ++it) {
        const int k0 = it * BK;
        // prefetch next B tile into the alternate buffer
        if (wave == 0 && it + 1 < nIter)
            tdm_load_2d_bf16((it + 1) & 1 ? bOff1 : bOff0,
                             bBase + (unsigned long long)(k0 + BK) * 2, BK, BN, H);

        // stage A: gathered token rows, f32 -> bf16 (overlaps B DMA)
        {
            const int tok = stok[rA];
            unsigned int pk[8];
            if (tok >= 0) {
                const float* src = x + (size_t)tok * H + k0 + halfA * 16;
#pragma unroll
                for (int j = 0; j < 8; ++j)
                    pk[j] = (unsigned int)f2bf(src[2 * j]) |
                            ((unsigned int)f2bf(src[2 * j + 1]) << 16);
            } else {
#pragma unroll
                for (int j = 0; j < 8; ++j) pk[j] = 0u;
            }
            unsigned int* dst = (unsigned int*)&As[rA * BK + halfA * 16];
#pragma unroll
            for (int j = 0; j < 8; ++j) dst[j] = pk[j];
        }
        if (wave == 0) {
            if (it + 1 < nIter) __builtin_amdgcn_s_wait_tensorcnt(1); // tile it done
            else                __builtin_amdgcn_s_wait_tensorcnt(0);
        }
        __syncthreads();

        const unsigned short* Bcur = Bs[it & 1];
        Frag16 a[2], b[4];
#pragma unroll
        for (int tm = 0; tm < 2; ++tm) {
            const int row = (waveM * 32 + tm * 16 + lm) * BK;
            a[tm].u[0] = *(const u32x4*)&As[row + hf];
            a[tm].u[1] = *(const u32x4*)&As[row + 16 + hf];
        }
#pragma unroll
        for (int tn = 0; tn < 4; ++tn) {
            const int col = (waveN * 64 + tn * 16 + lm) * BK;
            b[tn].u[0] = *(const u32x4*)&Bcur[col + hf];
            b[tn].u[1] = *(const u32x4*)&Bcur[col + 16 + hf];
        }
#pragma unroll
        for (int tm = 0; tm < 2; ++tm)
#pragma unroll
            for (int tn = 0; tn < 4; ++tn)
                acc[tm][tn] = __builtin_amdgcn_wmma_f32_16x16x32_bf16(
                    false, a[tm].v, false, b[tn].v, (short)0, acc[tm][tn],
                    false, false);
        __syncthreads();
    }

#pragma unroll
    for (int tm = 0; tm < 2; ++tm) {
#pragma unroll
        for (int r = 0; r < 8; ++r) {
            const int row = m0 + waveM * 32 + tm * 16 + r + (lane >> 4) * 8;
#pragma unroll
            for (int tn = 0; tn < 4; ++tn) {
                const int col = n0 + waveN * 64 + tn * 16 + lm;
                float v = acc[tm][tn][r];
                v = v > 0.f ? v : 0.f;
                hbuf[((size_t)e * CAP + row) * I + col] = f2bf(v);
            }
        }
    }
}

// ---------------------------------------------------------------------------
// GEMM2: y[e] = h[e] @ W2[e].  A (h, bf16) and B (w2t, bf16) tiles are both
// double-buffered TDM copies: pair k+1 streams in while pair k is consumed
// (wait tensorcnt<=2 keeps one pair in flight; TENSORcnt completes in order).
// ---------------------------------------------------------------------------
__global__ __launch_bounds__(256) void moe_gemm2(
    const unsigned short* __restrict__ hbuf, const unsigned short* __restrict__ w2t,
    float* __restrict__ ybuf, int I, int H, int CAP)
{
    __shared__ __align__(16) unsigned short As[2][BM * BK];
    __shared__ __align__(16) unsigned short Bs[2][BN * BK];

    const int e  = blockIdx.z;
    const int n0 = blockIdx.x * BN;
    const int m0 = blockIdx.y * BM;
    const int tid = threadIdx.x;
    const int wave = tid >> 5, lane = tid & 31;
    const int waveM = wave & 3, waveN = wave >> 2;
    const int lm = lane & 15;
    const int hf = (lane >> 4) * 8;

    v8f acc[2][4];
#pragma unroll
    for (int i = 0; i < 2; ++i)
#pragma unroll
        for (int j = 0; j < 4; ++j) acc[i][j] = v8f{};

    const unsigned int aOff[2] = { lds_offset(As[0]), lds_offset(As[1]) };
    const unsigned int bOff[2] = { lds_offset(Bs[0]), lds_offset(Bs[1]) };
    const unsigned long long aBase =
        (unsigned long long)(uintptr_t)(hbuf + ((size_t)e * CAP + m0) * I);
    const unsigned long long bBase =
        (unsigned long long)(uintptr_t)(w2t + ((size_t)e * H + n0) * I);

    const int nIter = I / BK;
    if (wave == 0) {                               // prologue: pair 0
        tdm_load_2d_bf16(aOff[0], aBase, BK, BM, I);
        tdm_load_2d_bf16(bOff[0], bBase, BK, BN, I);
    }

    for (int it = 0; it < nIter; ++it) {
        if (wave == 0) {
            if (it + 1 < nIter) {                  // prefetch pair it+1
                const unsigned long long koff =
                    (unsigned long long)(it + 1) * BK * 2;
                const int nb = (it + 1) & 1;
                tdm_load_2d_bf16(aOff[nb], aBase + koff, BK, BM, I);
                tdm_load_2d_bf16(bOff[nb], bBase + koff, BK, BN, I);
                __builtin_amdgcn_s_wait_tensorcnt(2);   // pair it complete
            } else {
                __builtin_amdgcn_s_wait_tensorcnt(0);
            }
        }
        __syncthreads();

        const unsigned short* Acur = As[it & 1];
        const unsigned short* Bcur = Bs[it & 1];
        Frag16 a[2], b[4];
#pragma unroll
        for (int tm = 0; tm < 2; ++tm) {
            const int row = (waveM * 32 + tm * 16 + lm) * BK;
            a[tm].u[0] = *(const u32x4*)&Acur[row + hf];
            a[tm].u[1] = *(const u32x4*)&Acur[row + 16 + hf];
        }
#pragma unroll
        for (int tn = 0; tn < 4; ++tn) {
            const int col = (waveN * 64 + tn * 16 + lm) * BK;
            b[tn].u[0] = *(const u32x4*)&Bcur[col + hf];
            b[tn].u[1] = *(const u32x4*)&Bcur[col + 16 + hf];
        }
#pragma unroll
        for (int tm = 0; tm < 2; ++tm)
#pragma unroll
            for (int tn = 0; tn < 4; ++tn)
                acc[tm][tn] = __builtin_amdgcn_wmma_f32_16x16x32_bf16(
                    false, a[tm].v, false, b[tn].v, (short)0, acc[tm][tn],
                    false, false);
        __syncthreads();
    }

#pragma unroll
    for (int tm = 0; tm < 2; ++tm) {
#pragma unroll
        for (int r = 0; r < 8; ++r) {
            const int row = m0 + waveM * 32 + tm * 16 + r + (lane >> 4) * 8;
#pragma unroll
            for (int tn = 0; tn < 4; ++tn) {
                const int col = n0 + waveN * 64 + tn * 16 + lm;
                ybuf[((size_t)e * CAP + row) * H + col] = acc[tm][tn][r];
            }
        }
    }
}

// ---------------------------------------------------------------------------
// Phase 6: deterministic combine: out = x*noop_w + sum_k w_k * y[e_k][slot_k].
// ---------------------------------------------------------------------------
__global__ __launch_bounds__(256) void moe_combine(
    const float* __restrict__ x, const float* __restrict__ noop_w,
    const int* __restrict__ a_e, const float* __restrict__ a_w,
    const int* __restrict__ tok_slot, const float* __restrict__ ybuf,
    float* __restrict__ out, int H, int CAP)
{
    const size_t g = (size_t)blockIdx.x * blockDim.x + threadIdx.x;
    const size_t i = g * 4;
    const int t = (int)(i / H);
    const int h = (int)(i % H);

    const float4 xv = ((const float4*)x)[g];
    const float nw = noop_w[t];
    float r0 = xv.x * nw, r1 = xv.y * nw, r2 = xv.z * nw, r3 = xv.w * nw;

#pragma unroll
    for (int k = 0; k < 2; ++k) {
        const int s = tok_slot[2 * t + k];
        if (s >= 0) {
            const int e = a_e[2 * t + k];
            const float wv = a_w[2 * t + k];
            const float4 yv =
                *(const float4*)(ybuf + ((size_t)e * CAP + s) * H + h);
            r0 += wv * yv.x; r1 += wv * yv.y; r2 += wv * yv.z; r3 += wv * yv.w;
        }
    }
    ((float4*)out)[g] = make_float4(r0, r1, r2, r3);
}

// ---------------------------------------------------------------------------
extern "C" void kernel_launch(void* const* d_in, const int* in_sizes, int n_in,
                              void* d_out, int out_size, void* d_ws, size_t ws_size,
                              hipStream_t stream) {
    const float* x  = (const float*)d_in[0];
    const float* rw = (const float*)d_in[1];
    const float* rb = (const float*)d_in[2];
    const float* w1 = (const float*)d_in[3];
    const float* w2 = (const float*)d_in[4];
    float* out = (float*)d_out;

    const int H   = in_sizes[1] / 8;            // 1024
    const int N   = in_sizes[0] / H;            // 16384
    const int I   = in_sizes[3] / (7 * H);      // 2048
    const int CAP = (N + 7) / 8;                // 2048
    const int nRB = N / 8;

    auto alignup = [](size_t v) { return (v + 255) & ~(size_t)255; };
    char* p = (char*)d_ws;
    float* noop_w     = (float*)p;          p += alignup((size_t)N * 4);
    int*   a_e        = (int*)p;            p += alignup((size_t)2 * N * 4);
    float* a_w        = (float*)p;          p += alignup((size_t)2 * N * 4);
    int*   bcnt       = (int*)p;            p += alignup((size_t)nRB * 7 * 4);
    int*   boff       = (int*)p;            p += alignup((size_t)nRB * 7 * 4);
    int*   slot_token = (int*)p;            p += alignup((size_t)7 * CAP * 4);
    int*   tok_slot   = (int*)p;            p += alignup((size_t)2 * N * 4);
    unsigned short* w1t  = (unsigned short*)p; p += alignup((size_t)7 * H * I * 2);
    unsigned short* w2t  = (unsigned short*)p; p += alignup((size_t)7 * I * H * 2);
    unsigned short* hbuf = (unsigned short*)p; p += alignup((size_t)7 * CAP * I * 2);
    float* ybuf       = (float*)p;          p += alignup((size_t)7 * CAP * H * 4);

    // one-time weight convert+transpose (enables TDM-fed B tiles)
    moe_cvt_transpose<<<dim3(I / 32, H / 32, 7), 256, 0, stream>>>(w1, w1t, H, I);
    moe_cvt_transpose<<<dim3(H / 32, I / 32, 7), 256, 0, stream>>>(w2, w2t, I, H);

    moe_router<<<nRB, 256, 0, stream>>>(x, rw, rb, noop_w, a_e, a_w, bcnt, H);
    moe_scan_init<<<1, 256, 0, stream>>>(bcnt, boff, slot_token, nRB, CAP);
    moe_assign<<<nRB, 32, 0, stream>>>(a_e, a_w, boff, slot_token, tok_slot, CAP);
    moe_gemm1<<<dim3(I / BN, CAP / BM, 7), 256, 0, stream>>>(
        x, w1t, slot_token, hbuf, H, I, CAP);
    moe_gemm2<<<dim3(H / BN, CAP / BM, 7), 256, 0, stream>>>(
        hbuf, w2t, ybuf, I, H, CAP);
    moe_combine<<<(unsigned)(((size_t)N * H / 4) / 256), 256, 0, stream>>>(
        x, noop_w, a_e, a_w, tok_slot, ybuf, out, H, CAP);
}
